// PointerDecoder_65412351918622
// MI455X (gfx1250) — matrix operand order repaired and checked by reading, not compiled, not run on hardware
//
#include <hip/hip_runtime.h>
#include <hip/hip_bf16.h>

// ---------------------------------------------------------------------------
// Problem constants (from reference): B=256, N=256, E=256, H=512, steps=256
// ---------------------------------------------------------------------------
#define BB   256
#define NN   256
#define EE   256
#define HH   512
#define G4H  2048           // 4*H
#define KCAT 768            // E + H
#define STEPS 256

typedef __attribute__((ext_vector_type(16))) __bf16 v16bf;
typedef __attribute__((ext_vector_type(8)))  float  v8f;

union FragBF {
    v16bf v;
    uint4 q[2];
};

__device__ __forceinline__ float bf2f(unsigned v) {
    return __uint_as_float(v << 16);
}
__device__ __forceinline__ unsigned short f2bf(float f) {
    unsigned u = __float_as_uint(f);
    u += 0x7FFFu + ((u >> 16) & 1u);          // round-to-nearest-even
    return (unsigned short)(u >> 16);
}

// gfx1250 has a hardware V_TANH_F32 (TRANS op). Use it if the builtin exists.
#if defined(__has_builtin)
#if __has_builtin(__builtin_amdgcn_tanhf)
#define FAST_TANH(x) __builtin_amdgcn_tanhf(x)
#endif
#if __has_builtin(__builtin_amdgcn_rcpf)
#define FAST_RCP(x) __builtin_amdgcn_rcpf(x)
#endif
#endif
#ifndef FAST_TANH
#define FAST_TANH(x) tanhf(x)
#endif
#ifndef FAST_RCP
#define FAST_RCP(x) (1.0f / (x))
#endif

__device__ __forceinline__ float sigf(float x) {
    return FAST_RCP(1.0f + __expf(-x));
}
__device__ __forceinline__ float urand(unsigned a) {
    a ^= a >> 16; a *= 0x7FEB352Du;
    a ^= a >> 15; a *= 0x846CA68Bu;
    a ^= a >> 16;
    return (float)(a >> 8) * (1.0f / 16777216.0f) + 1.0e-7f;
}

// ---------------------------------------------------------------------------
// bf16 WMMA GEMM with LDS-staged A tile:
//   C(MxNcols) = A(MxK) * B^T (+ bias)
//   All 8 waves of a block share one 16-row A tile (same tm), so stage it in
//   LDS once (<=24KB) and read WMMA A-fragments via ds_load_b128.
//   B (row-major Ncols x K) streams from L2 per wave (distinct column tiles).
//   Grid sized exactly; requires (8*16) | tiles and same tm within a block.
// ---------------------------------------------------------------------------
__global__ __launch_bounds__(256)
void gemm_bf16_wmma(const unsigned short* __restrict__ A, int lda,
                    const unsigned short* __restrict__ Bw, int ldb,
                    float* __restrict__ Cf, unsigned short* __restrict__ Cbf,
                    const float* __restrict__ bias,
                    int n_tiles, int K, int ldc)
{
    __shared__ unsigned short sA[16 * KCAT];     // max K = 768 -> 24 KB

    const int wave = threadIdx.x >> 5;
    const int lane = threadIdx.x & 31;
    const int gid  = blockIdx.x * 8 + wave;
    const int tm   = gid / n_tiles;
    const int tn   = gid - tm * n_tiles;
    const int half = lane >> 4;
    const int l15  = lane & 15;

    // ---- stage A tile (16 x K bf16) into LDS, uint4 (8-elem) granularity ----
    {
        const int nvec = (16 * K) >> 3;          // # of 8-element chunks
        for (int i = threadIdx.x; i < nvec; i += 256) {
            const int ei  = i << 3;
            const int row = ei / K;
            const int off = ei - row * K;
            *(uint4*)(sA + row * K + off) =
                *(const uint4*)(A + (size_t)(tm * 16 + row) * lda + off);
        }
    }
    __syncthreads();

    const unsigned short* arow = sA + (size_t)l15 * K;
    const unsigned short* brow = Bw + (size_t)(tn * 16 + l15) * ldb;

    v8f acc = {};
    for (int k0 = 0; k0 < K; k0 += 32) {
        __builtin_prefetch(brow + k0 + 128, 0, 3);   // near-cache prefetch
        FragBF a, b;
        // A 16x32 bf16 fragment: lane (half,l15): K = half*8+[0..7], 16+half*8+[0..7]
        a.q[0] = *(const uint4*)(arow + k0 + half * 8);        // ds_load_b128
        a.q[1] = *(const uint4*)(arow + k0 + 16 + half * 8);
        // B 32x16 bf16 fragment: lane: N=l15, K = half*8+[0..7], 16+half*8+[0..7] (row of W)
        b.q[0] = *(const uint4*)(brow + k0 + half * 8);
        b.q[1] = *(const uint4*)(brow + k0 + 16 + half * 8);
        acc = __builtin_amdgcn_wmma_f32_16x16x32_bf16(
                  false, a.v, false, b.v, (short)0, acc, false, false);
    }

    const int col = tn * 16 + l15;
    const float bv = bias ? bias[col] : 0.0f;
#pragma unroll
    for (int r = 0; r < 8; ++r) {
        const int row = tm * 16 + half * 8 + r;     // C layout: M = r + half*8
        const float val = acc[r] + bv;
        if (Cf)  Cf [(size_t)row * ldc + col] = val;
        else     Cbf[(size_t)row * ldc + col] = f2bf(val);
    }
}

// ---------------------------------------------------------------------------
// One-time packing / conversion kernels
// ---------------------------------------------------------------------------
__global__ __launch_bounds__(256)
void cvt_f32_to_bf16(const float* __restrict__ src, unsigned short* __restrict__ dst, int n)
{
    int e = blockIdx.x * 256 + threadIdx.x;
    if (e < n) dst[e] = f2bf(src[e]);
}

__global__ __launch_bounds__(256)
void pack_wcat(const float* __restrict__ Wih, const float* __restrict__ Whh,
               unsigned short* __restrict__ Wcat)
{
    int e = blockIdx.x * 256 + threadIdx.x;      // 2048 * 768
    int col = e / KCAT, k = e - col * KCAT;
    float v = (k < EE) ? Wih[col * EE + k] : Whh[col * HH + (k - EE)];
    Wcat[e] = f2bf(v);
}

__global__ __launch_bounds__(256)
void add_bias(const float* __restrict__ bih, const float* __restrict__ bhh,
              float* __restrict__ bias)
{
    int e = blockIdx.x * 256 + threadIdx.x;      // 2048
    bias[e] = bih[e] + bhh[e];
}

__global__ __launch_bounds__(256)
void init_state(unsigned short* __restrict__ xcat, float* __restrict__ c,
                float* __restrict__ mask, float* __restrict__ outp)
{
    int e = blockIdx.x * 256 + threadIdx.x;      // grid covers 196608
    if (e < BB * KCAT)  xcat[e] = 0;             // start=0, h=0
    if (e < BB * HH)    c[e]    = 0.0f;          // c=0
    if (e < BB * NN)    mask[e] = 0.0f;          // visited=0
    if (e < BB)         outp[BB * STEPS + e] = 0.0f;   // logp_sum accumulator
}

// ---------------------------------------------------------------------------
// Fused LSTM elementwise update + q-projection WMMA GEMM.
//   Block tm handles batch rows [tm*16, tm*16+16):
//     phase 1: c,h update from gates; h (bf16) -> LDS (A layout) + xcat + c
//     phase 2: q[tile] = h @ Wq^T via WMMA, A-fragments from LDS.
//   Saves one kernel launch on the 256-step sequential critical path.
// ---------------------------------------------------------------------------
__global__ __launch_bounds__(256)
void lstm_qproj_fused(const float* __restrict__ gates,        // (B, 4H) fp32
                      float* __restrict__ c,                  // (B, H) fp32
                      unsigned short* __restrict__ xcat,      // (B, 768) bf16
                      const unsigned short* __restrict__ Wq_bf,// (H, H) bf16
                      float* __restrict__ qbuf)               // (B, H) fp32
{
    __shared__ unsigned short hA[16 * HH];       // 16 KB

    const int tm = blockIdx.x;                   // 0..15
    const int t  = threadIdx.x;

    // ---- phase 1: LSTM cell update for 16 rows x 512 cols ----
#pragma unroll
    for (int i = 0; i < 32; ++i) {
        const int e = i * 256 + t;               // 0..8191
        const int r = e >> 9;                    // row within tile
        const int h = e & (HH - 1);
        const int b = tm * 16 + r;
        const float* g = gates + (size_t)b * G4H;
        const float ig = sigf(g[h]);
        const float fg = sigf(g[HH + h]);
        const float gg = FAST_TANH(g[2 * HH + h]);
        const float og = sigf(g[3 * HH + h]);
        const float cn = fg * c[(size_t)b * HH + h] + ig * gg;
        const float hn = og * FAST_TANH(cn);
        c[(size_t)b * HH + h] = cn;
        const unsigned short hb = f2bf(hn);
        xcat[(size_t)b * KCAT + EE + h] = hb;    // for next step's gates GEMM
        hA[r * HH + h] = hb;
    }
    __syncthreads();

    // ---- phase 2: q = h @ Wq^T ; 32 column tiles over 8 waves (4 each) ----
    const int wave = t >> 5;
    const int lane = t & 31;
    const int half = lane >> 4;
    const int l15  = lane & 15;
    const unsigned short* arow = hA + (size_t)l15 * HH;

#pragma unroll
    for (int tt = 0; tt < 4; ++tt) {
        const int tn = wave * 4 + tt;
        const unsigned short* brow = Wq_bf + (size_t)(tn * 16 + l15) * HH;
        v8f acc = {};
        for (int k0 = 0; k0 < HH; k0 += 32) {
            FragBF a, b;
            a.q[0] = *(const uint4*)(arow + k0 + half * 8);      // ds_load_b128
            a.q[1] = *(const uint4*)(arow + k0 + 16 + half * 8);
            b.q[0] = *(const uint4*)(brow + k0 + half * 8);
            b.q[1] = *(const uint4*)(brow + k0 + 16 + half * 8);
            acc = __builtin_amdgcn_wmma_f32_16x16x32_bf16(
                      false, a.v, false, b.v, (short)0, acc, false, false);
        }
        const int col = tn * 16 + l15;
#pragma unroll
        for (int r = 0; r < 8; ++r) {
            const int row = tm * 16 + half * 8 + r;
            qbuf[(size_t)row * HH + col] = acc[r];
        }
    }
}

// ---------------------------------------------------------------------------
// Attention + masked softmax + Gumbel-max categorical sample + gather.
// One block per batch row b; thread t == candidate index n.
// k_enc is bf16 (67 MB -> resident in the 192 MB L2); each element is used
// exactly once per step by exactly one lane => direct b128 loads, no LDS.
// ---------------------------------------------------------------------------
__global__ __launch_bounds__(256)
void attention_step(const unsigned short* __restrict__ kenc,   // (B*N, H) bf16
                    const float* __restrict__ qrow,            // (B, H) fp32
                    const float* __restrict__ vvec,            // (H)
                    float* __restrict__ mask,                  // (B, N)
                    const unsigned short* __restrict__ encb,   // (B*N, E) bf16
                    unsigned short* __restrict__ xcat,         // (B, 768) bf16
                    float* __restrict__ outp,                  // d_out (float)
                    int step)
{
    __shared__ float qs[HH];
    __shared__ float vs[HH];
    __shared__ float rv[256];
    __shared__ int   ri[256];
    __shared__ int   sIdx;

    const int b = blockIdx.x;
    const int t = threadIdx.x;

    qs[t]       = qrow[b * HH + t];
    qs[t + 256] = qrow[b * HH + 256 + t];
    vs[t]       = vvec[t];
    vs[t + 256] = vvec[256 + t];
    __syncthreads();

    const int n = t;
    const unsigned short* krow = kenc + ((size_t)b * NN + n) * HH;
    float s = 0.0f;
    for (int h = 0; h < HH; h += 8) {
        uint4 pk = *(const uint4*)(krow + h);    // 8 bf16
        s += vs[h + 0] * FAST_TANH(qs[h + 0] + bf2f(pk.x & 0xFFFFu));
        s += vs[h + 1] * FAST_TANH(qs[h + 1] + bf2f(pk.x >> 16));
        s += vs[h + 2] * FAST_TANH(qs[h + 2] + bf2f(pk.y & 0xFFFFu));
        s += vs[h + 3] * FAST_TANH(qs[h + 3] + bf2f(pk.y >> 16));
        s += vs[h + 4] * FAST_TANH(qs[h + 4] + bf2f(pk.z & 0xFFFFu));
        s += vs[h + 5] * FAST_TANH(qs[h + 5] + bf2f(pk.z >> 16));
        s += vs[h + 6] * FAST_TANH(qs[h + 6] + bf2f(pk.w & 0xFFFFu));
        s += vs[h + 7] * FAST_TANH(qs[h + 7] + bf2f(pk.w >> 16));
    }
    s -= 1.0e6f * mask[b * NN + n];              // masked logits

    // block max
    rv[t] = s; __syncthreads();
    for (int st = 128; st > 0; st >>= 1) {
        if (t < st) rv[t] = fmaxf(rv[t], rv[t + st]);
        __syncthreads();
    }
    const float smax = rv[0]; __syncthreads();

    // block sum of exp
    const float e = __expf(s - smax);
    rv[t] = e; __syncthreads();
    for (int st = 128; st > 0; st >>= 1) {
        if (t < st) rv[t] += rv[t + st];
        __syncthreads();
    }
    const float ssum = rv[0]; __syncthreads();

    // Gumbel-max == categorical over masked logits
    const unsigned seed = ((unsigned)step << 16) | ((unsigned)b << 8) | (unsigned)n;
    const float u   = urand(seed);
    const float key = s - __logf(-__logf(u));
    rv[t] = key; ri[t] = t; __syncthreads();
    for (int st = 128; st > 0; st >>= 1) {
        if (t < st && rv[t + st] > rv[t]) { rv[t] = rv[t + st]; ri[t] = ri[t + st]; }
        __syncthreads();
    }
    if (t == 0) sIdx = ri[0];
    __syncthreads();
    const int idx = sIdx;

    if (t == idx) {
        const float prob = e * FAST_RCP(ssum);             // softmax prob of chosen
        outp[b * STEPS + step] = (float)idx;               // action_sequence
        outp[BB * STEPS + b]  += prob;                     // sum of probs (as reference)
        mask[b * NN + idx]     = 1.0f;                     // visited
    }
    // gather new start = encoder_output[b, idx]  (bf16, E=256)
    xcat[(size_t)b * KCAT + t] = encb[((size_t)b * NN + idx) * EE + t];
}

// ---------------------------------------------------------------------------
// Host-side orchestration (graph-capture safe: only kernel launches on stream)
// ---------------------------------------------------------------------------
static inline size_t align256(size_t x) { return (x + 255) & ~(size_t)255; }

extern "C" void kernel_launch(void* const* d_in, const int* in_sizes, int n_in,
                              void* d_out, int out_size, void* d_ws, size_t ws_size,
                              hipStream_t stream)
{
    (void)in_sizes; (void)n_in; (void)out_size; (void)ws_size;

    const float* enc  = (const float*)d_in[0];   // (B,N,E)
    const float* Wih  = (const float*)d_in[1];   // (4H,E)
    const float* Whh  = (const float*)d_in[2];   // (4H,H)
    const float* bih  = (const float*)d_in[3];   // (4H)
    const float* bhh  = (const float*)d_in[4];   // (4H)
    const float* Wq   = (const float*)d_in[5];   // (H,H)
    const float* Wk   = (const float*)d_in[6];   // (H,E)
    const float* vv   = (const float*)d_in[7];   // (H)
    float* outp = (float*)d_out;                 // [B*STEPS actions | B logp]

    // -------- workspace carve-up (bytes) --------
    char* ws = (char*)d_ws;
    size_t cur = 0;
    unsigned short* enc_bf  = (unsigned short*)(ws + cur); cur = align256(cur + (size_t)BB*NN*EE*2);
    unsigned short* kenc    = (unsigned short*)(ws + cur); cur = align256(cur + (size_t)BB*NN*HH*2);
    unsigned short* Wcat    = (unsigned short*)(ws + cur); cur = align256(cur + (size_t)G4H*KCAT*2);
    unsigned short* Wq_bf   = (unsigned short*)(ws + cur); cur = align256(cur + (size_t)HH*HH*2);
    unsigned short* Wk_bf   = (unsigned short*)(ws + cur); cur = align256(cur + (size_t)HH*EE*2);
    float*          bias    = (float*)(ws + cur);          cur = align256(cur + (size_t)G4H*4);
    float*          gates   = (float*)(ws + cur);          cur = align256(cur + (size_t)BB*G4H*4);
    float*          qbuf    = (float*)(ws + cur);          cur = align256(cur + (size_t)BB*HH*4);
    float*          cbuf    = (float*)(ws + cur);          cur = align256(cur + (size_t)BB*HH*4);
    unsigned short* xcat    = (unsigned short*)(ws + cur); cur = align256(cur + (size_t)BB*KCAT*2);
    float*          mask    = (float*)(ws + cur);          cur = align256(cur + (size_t)BB*NN*4);

    // -------- one-time packing --------
    cvt_f32_to_bf16<<<(BB*NN*EE)/256, 256, 0, stream>>>(enc, enc_bf, BB*NN*EE);
    pack_wcat      <<<(G4H*KCAT)/256, 256, 0, stream>>>(Wih, Whh, Wcat);
    cvt_f32_to_bf16<<<(HH*HH)/256,    256, 0, stream>>>(Wq, Wq_bf, HH*HH);
    cvt_f32_to_bf16<<<(HH*EE)/256,    256, 0, stream>>>(Wk, Wk_bf, HH*EE);
    add_bias       <<<G4H/256,        256, 0, stream>>>(bih, bhh, bias);
    init_state     <<<(BB*KCAT)/256,  256, 0, stream>>>(xcat, cbuf, mask, outp);

    // k_enc = enc @ Wk^T : (65536 x 512), K=256, bf16 output (L2 resident)
    {
        const int n_tiles = HH / 16;                     // 32
        const int tiles   = (BB*NN/16) * n_tiles;        // 131072
        gemm_bf16_wmma<<<tiles/8, 256, 0, stream>>>(
            enc_bf, EE, Wk_bf, EE, nullptr, kenc, nullptr, n_tiles, EE, HH);
    }

    // -------- 256 sequential decode steps (3 launches per step) --------
    for (int step = 0; step < STEPS; ++step) {
        // gates = [start|h] @ Wcat^T + bias : (256 x 2048), K=768
        {
            const int n_tiles = G4H / 16;                // 128
            const int tiles   = (BB/16) * n_tiles;       // 2048
            gemm_bf16_wmma<<<tiles/8, 256, 0, stream>>>(
                xcat, KCAT, Wcat, KCAT, gates, nullptr, bias, n_tiles, KCAT, G4H);
        }

        // fused: c,h update (h -> bf16 LDS + xcat) then q = h @ Wq^T
        lstm_qproj_fused<<<BB/16, 256, 0, stream>>>(gates, cbuf, xcat, Wq_bf, qbuf);

        // attention scores + masked softmax + sample + gather
        attention_step<<<BB, 256, 0, stream>>>(
            kenc, qbuf, vv, mask, enc_bf, xcat, outp, step);
    }
}
